// Attention_75857712382119
// MI455X (gfx1250) — matrix operand order repaired and checked by reading, not compiled, not run on hardware
//
#include <hip/hip_runtime.h>
#include <hip/hip_bf16.h>

// Problem constants
#define BATCH 512
#define HW    196
#define ENCD  2048
#define DECD  512
#define ATTND 512

#define MT        13        // ceil(196/16) M tiles
#define MPAD      208       // 13*16 padded rows
#define KC        64        // K chunk staged in LDS
#define NCHUNK    256       // N chunk (2 chunks cover ATTN=512)
#define NT_PER_WAVE 2       // N tiles owned per wave (8 waves * 2 * 16 = 256)

typedef __attribute__((ext_vector_type(16))) __bf16 bf16x16;
typedef __attribute__((ext_vector_type(8)))  float  floatx8;

union Frag16 { bf16x16 v; uint4 q[2]; };

__device__ __forceinline__ unsigned short f2bf(float x) {
  union { float f; unsigned int u; } c; c.f = x;
  unsigned int u = c.u;
  return (unsigned short)((u + 0x7FFFu + ((u >> 16) & 1u)) >> 16); // RNE
}

// ---- LDS layout (bytes), total 64320 < 65536 ----
// As  @ 0      : 208*64 bf16 = 26624   (aliases dh[512] f32 during stage 1)
// Bt  @ 26624  : 256*64 bf16 = 32768   (aliases red[256] / alpha[208] in stages 3/4)
// bias@ 59392  : 512 f32
// wv  @ 61440  : 512 f32
// log @ 63488  : 208 f32
#define AS_OFF    0
#define BT_OFF    26624
#define RED_OFF   26624
#define ALPHA_OFF (26624 + 4096)
#define BIAS_OFF  59392
#define WV_OFF    61440
#define LOG_OFF   63488
#define SMEM_BYTES 64320

__global__ __launch_bounds__(256)
void attn_fused_kernel(const float* __restrict__ enc_out,
                       const float* __restrict__ dec_h,
                       const float* __restrict__ W_enc,
                       const float* __restrict__ b_enc,
                       const float* __restrict__ W_dec,
                       const float* __restrict__ b_dec,
                       const float* __restrict__ W_v,
                       float* __restrict__ out_ctx,
                       float* __restrict__ out_alpha)
{
  __shared__ __align__(16) unsigned char smem[SMEM_BYTES];
  unsigned short* As    = (unsigned short*)(smem + AS_OFF);
  unsigned short* Bt    = (unsigned short*)(smem + BT_OFF);
  float* dh     = (float*)(smem + AS_OFF);
  float* red    = (float*)(smem + RED_OFF);
  float* alpha  = (float*)(smem + ALPHA_OFF);
  float* bias   = (float*)(smem + BIAS_OFF);
  float* wv     = (float*)(smem + WV_OFF);
  float* logits = (float*)(smem + LOG_OFF);

  const int t     = threadIdx.x;
  const int b     = blockIdx.x;
  const int w     = t >> 5;     // wave id 0..7 (wave32)
  const int l     = t & 31;     // lane
  const int lrow  = l & 15;
  const int lhalf = l >> 4;

  const float* encb = enc_out + (size_t)b * (HW * ENCD);

  // ================= Stage 1: bias[n] = b_enc + b_dec + dec_h . W_dec[:,n] ===
  dh[t]        = dec_h[b * DECD + t];
  dh[t + 256]  = dec_h[b * DECD + t + 256];
  wv[t]        = W_v[t];
  wv[t + 256]  = W_v[t + 256];
  if (t < MPAD) logits[t] = 0.f;
  __syncthreads();
  #pragma unroll 1
  for (int j = 0; j < 2; ++j) {
    const int n = t + (j << 8);
    float a = b_enc[n] + b_dec[n];
    #pragma unroll 4
    for (int d = 0; d < DECD; ++d)
      a = fmaf(dh[d], W_dec[d * ATTND + n], a);
    bias[n] = a;
  }
  // b_v deliberately omitted: softmax(logits + c) == softmax(logits), and
  // logits themselves are not an output.

  // ================= Stage 2: WMMA GEMM + fused relu/W_v reduction ==========
  for (int nchunk = 0; nchunk < 2; ++nchunk) {
    const int nbase = nchunk << 8;

    floatx8 acc[MT][NT_PER_WAVE];
    #pragma unroll
    for (int mt = 0; mt < MT; ++mt)
      #pragma unroll
      for (int nt = 0; nt < NT_PER_WAVE; ++nt)
        #pragma unroll
        for (int r = 0; r < 8; ++r)
          acc[mt][nt][r] = 0.f;

    for (int kc = 0; kc < ENCD / KC; ++kc) {
      __syncthreads();   // protect LDS from previous iteration / stage 1
      const int k0 = kc << 6;

      // A panel: enc_out[b, 0:196, k0:k0+64] -> bf16 As[m][k] (rows >=196 zero)
      #pragma unroll 1
      for (int idx = t; idx < MPAD * KC; idx += 256) {
        const int m = idx >> 6, k = idx & 63;
        const float vv = (m < HW) ? encb[m * ENCD + k0 + k] : 0.f;
        As[idx] = f2bf(vv);
      }
      // B panel: W_enc[k0:k0+64, nbase:nbase+256] -> bf16 Bt[n][k] (transposed)
      const float* wp = W_enc + (size_t)k0 * ATTND + nbase;
      #pragma unroll 1
      for (int idx = t; idx < KC * NCHUNK; idx += 256) {
        const int k = idx >> 8, n = idx & 255;
        Bt[n * KC + k] = f2bf(wp[k * ATTND + n]);
      }
      if (kc + 1 < ENCD / KC)
        __builtin_prefetch(encb + ((kc + 1) << 6) + (t << 4), 0, 0); // global_prefetch_b8
      __syncthreads();

      const unsigned char* AsB = (const unsigned char*)As;
      const unsigned char* BtB = (const unsigned char*)Bt;
      #pragma unroll
      for (int ks = 0; ks < 2; ++ks) {
        // B fragments: lane holds column n = lrow of its tile, q0: K = ks*32 + lhalf*8 + [0..7], q1: +16
        Frag16 bfr[NT_PER_WAVE];
        #pragma unroll
        for (int nt = 0; nt < NT_PER_WAVE; ++nt) {
          const int nl = (((w << 1) + nt) << 4) | lrow;
          const unsigned char* bp = BtB + nl * 128 + (ks << 6) + (lhalf << 4);
          bfr[nt].q[0] = *(const uint4*)(bp);
          bfr[nt].q[1] = *(const uint4*)(bp + 32);
        }
        #pragma unroll
        for (int mt = 0; mt < MT; ++mt) {
          // A fragment: lane holds row m = mt*16 + lrow,
          // q0: K = ks*32 + lhalf*8 + [0..7], q1: +16
          Frag16 af;
          const unsigned char* ap =
              AsB + ((mt << 4) + lrow) * 128 + (ks << 6) + (lhalf << 4);
          af.q[0] = *(const uint4*)(ap);
          af.q[1] = *(const uint4*)(ap + 32);
          #pragma unroll
          for (int nt = 0; nt < NT_PER_WAVE; ++nt)
            acc[mt][nt] = __builtin_amdgcn_wmma_f32_16x16x32_bf16(
                false, af.v, false, bfr[nt].v, (short)0, acc[mt][nt],
                false, false);
        }
      }
    }

    // Fused epilogue: logits[m] += sum_n relu(score[m,n]) * W_v[n]
    // C layout: VGPR r, lanes 0-15 -> (M=r, N=lane); lanes 16-31 -> (M=8+r, N=lane-16)
    #pragma unroll
    for (int nt = 0; nt < NT_PER_WAVE; ++nt) {
      const int ng = nbase + ((((w << 1) + nt) << 4) | lrow);
      const float bv = bias[ng];
      const float wf = wv[ng];
      #pragma unroll
      for (int mt = 0; mt < MT; ++mt) {
        #pragma unroll
        for (int r = 0; r < 8; ++r) {
          float sres = fmaxf(acc[mt][nt][r] + bv, 0.f) * wf;
          sres += __shfl_xor(sres, 1);
          sres += __shfl_xor(sres, 2);
          sres += __shfl_xor(sres, 4);
          sres += __shfl_xor(sres, 8);
          if (lrow == 0)
            atomicAdd(&logits[(mt << 4) + (lhalf << 3) + r], sres); // ds_add_f32
        }
      }
    }
  }

  // ================= Stage 3: softmax over HW=196 ===========================
  __syncthreads();
  const float lv = (t < HW) ? logits[t] : -3.4e38f;
  red[t] = lv;
  __syncthreads();
  for (int s = 128; s > 0; s >>= 1) {
    if (t < s) red[t] = fmaxf(red[t], red[t + s]);
    __syncthreads();
  }
  const float mx = red[0];
  __syncthreads();
  const float ev = (t < HW) ? __expf(lv - mx) : 0.f;
  red[t] = ev;
  __syncthreads();
  for (int s = 128; s > 0; s >>= 1) {
    if (t < s) red[t] += red[t + s];
    __syncthreads();
  }
  const float inv = 1.f / red[0];
  __syncthreads();
  if (t < HW) {
    const float a = ev * inv;
    alpha[t] = a;
    out_alpha[b * HW + t] = a;
  }
  __syncthreads();

  // ================= Stage 4: context = alpha^T . enc_out[b] ================
  float cacc[8];
  #pragma unroll
  for (int i = 0; i < 8; ++i) cacc[i] = 0.f;
  #pragma unroll 1
  for (int s = 0; s < HW; ++s) {
    const float a = alpha[s];
    const float* row = encb + s * ENCD;
    #pragma unroll
    for (int i = 0; i < 8; ++i)
      cacc[i] = fmaf(a, row[t + (i << 8)], cacc[i]);
  }
  float* ob = out_ctx + (size_t)b * ENCD;
  #pragma unroll
  for (int i = 0; i < 8; ++i) ob[t + (i << 8)] = cacc[i];
}

extern "C" void kernel_launch(void* const* d_in, const int* in_sizes, int n_in,
                              void* d_out, int out_size, void* d_ws, size_t ws_size,
                              hipStream_t stream) {
  (void)in_sizes; (void)n_in; (void)out_size; (void)d_ws; (void)ws_size;
  const float* enc_out = (const float*)d_in[0];
  const float* dec_h   = (const float*)d_in[1];
  const float* W_enc   = (const float*)d_in[2];
  const float* b_enc   = (const float*)d_in[3];
  const float* W_dec   = (const float*)d_in[4];
  const float* b_dec   = (const float*)d_in[5];
  const float* W_v     = (const float*)d_in[6];
  // d_in[7] = b_v: unused (softmax shift-invariance; logits not an output)

  float* out_ctx   = (float*)d_out;
  float* out_alpha = out_ctx + (size_t)BATCH * ENCD;

  attn_fused_kernel<<<BATCH, 256, 0, stream>>>(
      enc_out, dec_h, W_enc, b_enc, W_dec, b_dec, W_v, out_ctx, out_alpha);
}
